// GCNLayer_75703093559638
// MI455X (gfx1250) — matrix operand order repaired and checked by reading, not compiled, not run on hardware
//
#include <hip/hip_runtime.h>
#include <hip/hip_bf16.h>

// ---------------------------------------------------------------------------
// GCN layer for MI455X (gfx1250, wave32, WMMA bf16 path)
//   Out = dinv[n] * ( (A @ Y)[n,o] + Y[n,o] ) + bias[o]
//   Y   = (dinv ⊙ X) @ W^T     (folds the linear layer through the A-GEMM)
//   dinv[n] = rsqrt(1 + sum_m A[n,m])
// ---------------------------------------------------------------------------

typedef __attribute__((ext_vector_type(16))) __bf16 v16bf;
typedef __attribute__((ext_vector_type(8)))  float  v8f;
typedef int v4i __attribute__((__vector_size__(16)));   // matches builtin param type

#if defined(__gfx1250__) && __has_builtin(__builtin_amdgcn_global_load_async_to_lds_b128)
#define USE_ASYNC_LDS 1
#else
#define USE_ASYNC_LDS 0
#endif

union BF16x16 {
    v16bf  v;
    __bf16 h[16];
    uint4  q[2];
};

__device__ __forceinline__ void pack16(BF16x16& d, const float4& a0, const float4& a1,
                                       const float4& a2, const float4& a3, float s) {
    // native fptrunc -> v_cvt_pk_bf16_f32 (RNE)
    d.h[ 0] = (__bf16)(s * a0.x); d.h[ 1] = (__bf16)(s * a0.y);
    d.h[ 2] = (__bf16)(s * a0.z); d.h[ 3] = (__bf16)(s * a0.w);
    d.h[ 4] = (__bf16)(s * a1.x); d.h[ 5] = (__bf16)(s * a1.y);
    d.h[ 6] = (__bf16)(s * a1.z); d.h[ 7] = (__bf16)(s * a1.w);
    d.h[ 8] = (__bf16)(s * a2.x); d.h[ 9] = (__bf16)(s * a2.y);
    d.h[10] = (__bf16)(s * a2.z); d.h[11] = (__bf16)(s * a2.w);
    d.h[12] = (__bf16)(s * a3.x); d.h[13] = (__bf16)(s * a3.y);
    d.h[14] = (__bf16)(s * a3.z); d.h[15] = (__bf16)(s * a3.w);
}

#define NB   4
#define NN   4096
#define ND   128
// Y fragment buffer: per batch 128 k-chunks * 8 o-tiles * 32 lanes * 16 elems (bf16)
#define YFRAG_PER_B (128 * 8 * 32 * 16)   // 524288 bf16 = 1 MB

// ---------------------------------------------------------------------------
// Kernel 1: degree + dinv.  One 256-thread block per (b,n) row of A.
// ---------------------------------------------------------------------------
__global__ __launch_bounds__(256) void gcn_deg(const float* __restrict__ A,
                                               float* __restrict__ dinv) {
    __shared__ float red[256];
    const size_t rowid = blockIdx.x;                 // b*4096 + n
    const float* ar = A + rowid * NN;
    float s = 0.f;
    #pragma unroll 4
    for (int i = threadIdx.x; i < NN; i += 256) s += ar[i];
    red[threadIdx.x] = s;
    __syncthreads();
    for (int off = 128; off > 0; off >>= 1) {
        if ((int)threadIdx.x < off) red[threadIdx.x] += red[threadIdx.x + off];
        __syncthreads();
    }
    if (threadIdx.x == 0) {
        float deg = red[0] + 1.0f;                   // + self loop
        dinv[rowid] = (deg > 0.f) ? rsqrtf(deg) : 0.f;
    }
}

// ---------------------------------------------------------------------------
// Kernel 2: Y = (dinv ⊙ X) @ W^T, emitted as bf16 B-fragments.
// One wave per 16-row tile of Y. 4 waves / block, 256 blocks.
// ---------------------------------------------------------------------------
__global__ __launch_bounds__(128) void gcn_prep_y(const float* __restrict__ X,
                                                  const float* __restrict__ W,
                                                  const float* __restrict__ dinv,
                                                  __bf16* __restrict__ Yfrag) {
    const int tile = blockIdx.x * 4 + (threadIdx.x >> 5);   // 0..1023
    const int b    = tile >> 8;
    const int n0   = (tile & 255) * 16;
    const int lane = threadIdx.x & 31;
    const int row  = lane & 15;
    const int lhi  = lane >> 4;

    const float dv = dinv[b * NN + n0 + row];
    const float* xrow = X + ((size_t)(b * NN + n0 + row)) * ND + 8 * lhi;

    // A-fragments of Xs: lane l holds row (l&15), k = 32c + 8*(l>>4) + {0..7, 16..23}
    BF16x16 af[4];
    #pragma unroll
    for (int c = 0; c < 4; ++c) {
        const float* ap = xrow + 32 * c;
        float4 a0 = *(const float4*)(ap);
        float4 a1 = *(const float4*)(ap + 4);
        float4 a2 = *(const float4*)(ap + 16);
        float4 a3 = *(const float4*)(ap + 20);
        pack16(af[c], a0, a1, a2, a3, dv);
    }

    const int p  = (n0 >> 4) & 1;
    const int co = n0 >> 5;
    __bf16* Yb = Yfrag + (size_t)b * YFRAG_PER_B;
    const int lp = row + 16 * p;

    #pragma unroll
    for (int t = 0; t < 8; ++t) {
        v8f acc = {0.f, 0.f, 0.f, 0.f, 0.f, 0.f, 0.f, 0.f};
        #pragma unroll
        for (int c = 0; c < 4; ++c) {
            // B-fragment of W^T: lane l holds col o=16t+(l&15), k=d=32c+8*(l>>4)+{0..7, 16..23}
            const float* wp = W + (size_t)(t * 16 + row) * ND + 32 * c + 8 * lhi;
            float4 w0 = *(const float4*)(wp);
            float4 w1 = *(const float4*)(wp + 4);
            float4 w2 = *(const float4*)(wp + 16);
            float4 w3 = *(const float4*)(wp + 20);
            BF16x16 bfv;
            pack16(bfv, w0, w1, w2, w3, 1.0f);
            acc = __builtin_amdgcn_wmma_f32_16x16x32_bf16(
                false, af[c].v, false, bfv.v, (short)0, acc, false, false);
        }
        // scatter-store into B-fragment layout for the main GEMM
        #pragma unroll
        for (int v = 0; v < 8; ++v) {
            int e = v + 8 * lhi;
            Yb[((size_t)(co * 8 + t) * 32 + lp) * 16 + e] = (__bf16)acc[v];
        }
    }
}

// ---------------------------------------------------------------------------
// Kernel 3: main GEMM  acc = A @ Y  (bf16 WMMA, fp32 accumulate) + epilogue.
// 4 waves / block, each wave: 16 rows x 128 cols, K = 4096.
// Y fragments staged 4 k-chunks (32 KB) at a time into LDS via
// GLOBAL_LOAD_ASYNC_TO_LDS_B128 (ASYNCcnt), falling back to a VGPR copy.
// ---------------------------------------------------------------------------
__global__ __launch_bounds__(128) void gcn_gemm(const float* __restrict__ A,
                                                const __bf16* __restrict__ Yfrag,
                                                const float* __restrict__ dinv,
                                                const float* __restrict__ bias,
                                                float* __restrict__ out) {
    __shared__ __align__(16) __bf16 lds_y[4 * 8 * 32 * 16];  // 32 KB

    const int b    = blockIdx.x >> 6;          // 64 row-blocks per batch
    const int rb   = blockIdx.x & 63;
    const int wave = threadIdx.x >> 5;
    const int lane = threadIdx.x & 31;
    const int n0   = rb * 64 + wave * 16;
    const int row  = lane & 15;
    const int lhi  = lane >> 4;

    const float* Ab = A + (size_t)b * NN * NN;
    const __bf16* Yb = Yfrag + (size_t)b * YFRAG_PER_B;
    const float* arow = Ab + (size_t)(n0 + row) * NN + 16 * lhi;

    v8f acc[8];
    #pragma unroll
    for (int t = 0; t < 8; ++t) acc[t] = (v8f){0.f, 0.f, 0.f, 0.f, 0.f, 0.f, 0.f, 0.f};

    for (int c0 = 0; c0 < 128; c0 += 4) {
        __syncthreads();                        // previous slab fully consumed
        {   // stage 4 contiguous k-chunks (32 KB) of Y fragments
            const uint4* src = (const uint4*)(Yb + (size_t)c0 * 4096);
            uint4* dst = (uint4*)lds_y;
#if USE_ASYNC_LDS
            #pragma unroll 4
            for (int i = threadIdx.x; i < 2048; i += 128) {
                __builtin_amdgcn_global_load_async_to_lds_b128(
                    (__attribute__((address_space(1))) v4i*)(src + i),
                    (__attribute__((address_space(3))) v4i*)(dst + i),
                    0, 0);
            }
            asm volatile("s_wait_asynccnt 0x0" ::: "memory");
#else
            #pragma unroll 4
            for (int i = threadIdx.x; i < 2048; i += 128) dst[i] = src[i];
#endif
        }
        __syncthreads();

        #pragma unroll
        for (int cc = 0; cc < 4; ++cc) {
            const int kb = (c0 + cc) * 32;
            // A-fragment: rows n0..n0+15, k = kb + 16*(l>>4) + {0..15}
            const float* ap = arow + kb;
            float4 a0 = *(const float4*)(ap);
            float4 a1 = *(const float4*)(ap + 4);
            float4 a2 = *(const float4*)(ap + 8);
            float4 a3 = *(const float4*)(ap + 12);
            BF16x16 af;
            pack16(af, a0, a1, a2, a3, 1.0f);

            #pragma unroll
            for (int t = 0; t < 8; ++t) {
                const uint4* lp = (const uint4*)&lds_y[((cc * 8 + t) * 32 + lane) * 16];
                BF16x16 bfv;
                bfv.q[0] = lp[0];
                bfv.q[1] = lp[1];
                acc[t] = __builtin_amdgcn_wmma_f32_16x16x32_bf16(
                    false, af.v, false, bfv.v, (short)0, acc[t], false, false);
            }
        }
    }

    // Epilogue: Out[n,o] = dinv[n] * (acc + Y[n,o]) + bias[o]
    float dn[8];
    #pragma unroll
    for (int v = 0; v < 8; ++v) dn[v] = dinv[b * NN + n0 + v + 8 * lhi];

    const int p  = (n0 >> 4) & 1;
    const int co = n0 >> 5;
    const int lp = row + 16 * p;

    #pragma unroll
    for (int t = 0; t < 8; ++t) {
        const float bs = bias[t * 16 + row];
        #pragma unroll
        for (int v = 0; v < 8; ++v) {
            const int e = v + 8 * lhi;
            const float y = (float)Yb[((size_t)(co * 8 + t) * 32 + lp) * 16 + e];
            const float val = dn[v] * (acc[t][v] + y) + bs;
            const size_t n = (size_t)(b * NN + n0 + v + 8 * lhi);
            out[n * ND + t * 16 + row] = val;
        }
    }
}

// ---------------------------------------------------------------------------
extern "C" void kernel_launch(void* const* d_in, const int* in_sizes, int n_in,
                              void* d_out, int out_size, void* d_ws, size_t ws_size,
                              hipStream_t stream) {
    const float* X    = (const float*)d_in[0];   // [4,4096,128]
    const float* A    = (const float*)d_in[1];   // [4,4096,4096]
    const float* W    = (const float*)d_in[2];   // [128,128]
    const float* bias = (const float*)d_in[3];   // [128]
    float* out = (float*)d_out;                  // [4,4096,128]

    float* dinv = (float*)d_ws;                                  // 64 KB
    __bf16* Yfrag = (__bf16*)((char*)d_ws + 65536);              // 4 MB

    gcn_deg   <<<NB * NN, 256, 0, stream>>>(A, dinv);
    gcn_prep_y<<<256,     128, 0, stream>>>(X, W, dinv, Yfrag);
    gcn_gemm  <<<256,     128, 0, stream>>>(A, Yfrag, dinv, bias, out);
}
